// LinearDiffusion_38182259261697
// MI455X (gfx1250) — compile-verified, matrix operand order checked by reading, not run on hardware
//
#include <hip/hip_runtime.h>

// ---------------------------------------------------------------------------
// LinearDiffusion for MI455X (gfx1250, wave32, WMMA + async global->LDS copy)
//   A = scatter(e) row-normalized (f32 atomics -> f16)
//   out = A@h + sum_{i=1..5} (A^(2^i))@h / i!
//   powers via repeated squaring with v_wmma_f32_16x16x32_f16;
//   A-tiles staged with global_load_async_to_lds_b128 (ASYNCcnt).
// ---------------------------------------------------------------------------

#define N_NODES 8192
#define DEGREE  32
#define N_EDGES (N_NODES * DEGREE)
#define D_FEAT  256

typedef __attribute__((ext_vector_type(16))) _Float16 v16h;
typedef __attribute__((ext_vector_type(8)))  float    v8f;

// ---------------- small utility kernels ----------------

__global__ void k_zero_f32(float* __restrict__ p, long n) {
  long i = (long)blockIdx.x * blockDim.x + threadIdx.x;
  long stride = (long)gridDim.x * blockDim.x;
  for (; i < n; i += stride) p[i] = 0.0f;
}

__global__ void k_scatter(const int* __restrict__ src, const int* __restrict__ dst,
                          const float* __restrict__ e, float* __restrict__ A) {
  int i = blockIdx.x * blockDim.x + threadIdx.x;
  if (i < N_EDGES) {
    long idx = (long)src[i] * N_NODES + (long)dst[i];
    atomicAdd(&A[idx], e[i]);
  }
}

// one 256-thread block per row: row-sum reduce, then normalized f16 row
__global__ void k_normalize(const float* __restrict__ A, _Float16* __restrict__ P) {
  __shared__ float red[256];
  const int r = blockIdx.x;
  const float* row = A + (long)r * N_NODES;
  float s = 0.0f;
  for (int c = threadIdx.x; c < N_NODES; c += 256) s += row[c];
  red[threadIdx.x] = s;
  __syncthreads();
  for (int off = 128; off > 0; off >>= 1) {
    if (threadIdx.x < off) red[threadIdx.x] += red[threadIdx.x + off];
    __syncthreads();
  }
  const float inv = 1.0f / red[0];
  _Float16* prow = P + (long)r * N_NODES;
  for (int c = threadIdx.x; c < N_NODES; c += 256)
    prow[c] = (_Float16)(row[c] * inv);
}

__global__ void k_f32_to_f16(const float* __restrict__ x, _Float16* __restrict__ y, long n) {
  long i = (long)blockIdx.x * blockDim.x + threadIdx.x;
  long stride = (long)gridDim.x * blockDim.x;
  for (; i < n; i += stride) y[i] = (_Float16)x[i];
}

// ---------------- async global -> LDS copy (CDNA5, ASYNCcnt) ----------------
// copies 16 bytes from global (base + 32-bit offset) straight into LDS, no VGPR data
__device__ __forceinline__ void async_copy_b128(unsigned lds_addr, unsigned goff,
                                                const void* base) {
  asm volatile("global_load_async_to_lds_b128 %0, %1, %2"
               :: "v"(lds_addr), "v"(goff), "s"(base)
               : "memory");
}
__device__ __forceinline__ void wait_asynccnt0() {
  asm volatile("s_wait_asynccnt 0x0" ::: "memory");
}

// ---------------- WMMA GEMM ----------------
// C[M x Nn] = A[M x Kd] * B[Kd x Nn]   (row-major, f16 in, f32 accumulate)
// HALF_OUT=1 : C written as f16 (matrix squaring)
// HALF_OUT=0 : C is f32, out = accum ? out + scale*C : scale*C (feature diffuse)

#define BM  128
#define BN  128
#define BK  32
#define LDT 40      // BK + 8 pad halfs; 80B row stride (16B aligned, conflict-breaking)

template <bool HALF_OUT>
__global__ __launch_bounds__(256)
void k_gemm_f16(const _Float16* __restrict__ A, const _Float16* __restrict__ B,
                void* __restrict__ Cout, int Nn, int Kd, float scale, int accum) {
  __shared__ __align__(16) _Float16 sA[2][BM][LDT];
  __shared__ __align__(16) _Float16 sB[2][BN][LDT];   // B stored transposed: [n][k]

  const int tid  = threadIdx.x;
  const int lane = tid & 31;
  const int wave = tid >> 5;
  const int wm0  = (wave >> 1) * 32;   // 4 waves tile M
  const int wn0  = (wave & 1) * 64;    // 2 waves tile N
  const int m0   = blockIdx.y * BM;
  const int n0   = blockIdx.x * BN;

  const int lrow = lane & 15;          // row (A) / col (B) within 16x16 tile
  const int kb   = (lane >> 4) * 8;    // K sub-base: 0 or 8 (ISA 16-bit frag layout)

  // ---- A tile: async copy addressing (2 x b128 chunks per thread) ----
  const int arow = tid >> 2;                       // rows 0..63 (chunk0), +64 (chunk1)
  const int ako  = (tid & 3) * 8;                  // half offset within row
  const unsigned ldsA0   = (unsigned)(uintptr_t)(&sA[0][arow][ako]);
  const unsigned ldsA1   = (unsigned)(uintptr_t)(&sA[0][arow + 64][ako]);
  const unsigned sAbytes = (unsigned)sizeof(sA[0]);          // buffer stride
  unsigned offA0 = (unsigned)(((long)(m0 + arow) * Kd + ako) * 2);
  unsigned offA1 = offA0 + (unsigned)((long)64 * Kd * 2);

  // ---- B tile: 2 adjacent K rows x 8 cols per thread, transposed store ----
  const int bk2 = (tid >> 4) * 2;                  // even K row 0..30
  const int bno = (tid & 15) * 8;                  // col chunk
  const _Float16* gB = B + (long)bk2 * Nn + n0 + bno;

  v8f acc[2][4];
#pragma unroll
  for (int i = 0; i < 2; ++i)
#pragma unroll
    for (int j = 0; j < 4; ++j)
#pragma unroll
      for (int v = 0; v < 8; ++v) acc[i][j][v] = 0.0f;

  auto storeB = [&](int buf, uint4 r0, uint4 r1) {
    const _Float16* h0 = (const _Float16*)&r0;
    const _Float16* h1 = (const _Float16*)&r1;
#pragma unroll
    for (int j = 0; j < 8; ++j) {                 // packed {k, k+1} dword store
      union { _Float16 h[2]; unsigned u; } pk;
      pk.h[0] = h0[j];
      pk.h[1] = h1[j];
      *(unsigned*)(&sB[buf][bno + j][bk2]) = pk.u;
    }
  };

  union FH { uint4 q[2]; v16h v; };

  const _Float16* aFragBase = &sA[0][wm0 + lrow][kb];
  const _Float16* bFragBase = &sB[0][wn0 + lrow][kb];

  auto compute = [&](int buf) {
    const _Float16* ab = aFragBase + buf * (BM * LDT);
    const _Float16* bb = bFragBase + buf * (BN * LDT);
    FH af[2], bf[4];
#pragma unroll
    for (int i = 0; i < 2; ++i) {
      af[i].q[0] = *(const uint4*)(ab + i * (16 * LDT));        // K = kb..kb+7
      af[i].q[1] = *(const uint4*)(ab + i * (16 * LDT) + 16);   // K = kb+16..kb+23
    }
#pragma unroll
    for (int j = 0; j < 4; ++j) {
      bf[j].q[0] = *(const uint4*)(bb + j * (16 * LDT));
      bf[j].q[1] = *(const uint4*)(bb + j * (16 * LDT) + 16);
    }
#pragma unroll
    for (int i = 0; i < 2; ++i)
#pragma unroll
      for (int j = 0; j < 4; ++j)
        acc[i][j] = __builtin_amdgcn_wmma_f32_16x16x32_f16(
            false, af[i].v, false, bf[j].v, (short)0, acc[i][j], false, false);
  };

  // ---- prologue: tile 0 ----
  async_copy_b128(ldsA0, offA0, A);
  async_copy_b128(ldsA1, offA1, A);
  {
    uint4 r0 = *(const uint4*)gB;
    uint4 r1 = *(const uint4*)(gB + Nn);
    storeB(0, r0, r1);
  }
  wait_asynccnt0();
  __syncthreads();

  const int nk = Kd / BK;
  for (int t = 0; t < nk; ++t) {
    const int buf = t & 1;
    uint4 r0, r1;
    if (t + 1 < nk) {
      offA0 += BK * 2;                          // advance one K-tile (bytes)
      offA1 += BK * 2;
      const unsigned bofs = (unsigned)(buf ^ 1) * sAbytes;
      async_copy_b128(ldsA0 + bofs, offA0, A);  // next A tile, no VGPR staging
      async_copy_b128(ldsA1 + bofs, offA1, A);
      gB += (long)BK * Nn;
      r0 = *(const uint4*)gB;                   // issue B loads before compute
      r1 = *(const uint4*)(gB + Nn);
    }
    compute(buf);
    if (t + 1 < nk) storeB(buf ^ 1, r0, r1);
    wait_asynccnt0();
    __syncthreads();
  }

  // Epilogue: C/D layout = 8 VGPRs; lanes 0-15 -> rows v (N=lane), 16-31 -> rows v+8
  const int colg  = lane & 15;
  const int rbase = (lane < 16) ? 0 : 8;
#pragma unroll
  for (int i = 0; i < 2; ++i) {
#pragma unroll
    for (int j = 0; j < 4; ++j) {
      const int gm = m0 + wm0 + i * 16 + rbase;
      const int gn = n0 + wn0 + j * 16 + colg;
#pragma unroll
      for (int v = 0; v < 8; ++v) {
        const float val = acc[i][j][v];
        const long idx = (long)(gm + v) * Nn + gn;
        if (HALF_OUT) {
          ((_Float16*)Cout)[idx] = (_Float16)val;
        } else {
          float* o = (float*)Cout;
          o[idx] = accum ? (o[idx] + scale * val) : (scale * val);
        }
      }
    }
  }
}

// ---------------- launch ----------------

extern "C" void kernel_launch(void* const* d_in, const int* in_sizes, int n_in,
                              void* d_out, int out_size, void* d_ws, size_t ws_size,
                              hipStream_t stream) {
  (void)in_sizes; (void)n_in; (void)out_size; (void)ws_size;

  const int*   src = (const int*)d_in[0];
  const int*   dst = (const int*)d_in[1];
  const float* e   = (const float*)d_in[2];
  const float* h   = (const float*)d_in[3];
  float*       out = (float*)d_out;

  char* ws = (char*)d_ws;
  const size_t nA = (size_t)N_NODES * N_NODES;
  float*    Af  = (float*)ws;                            // 256 MB scatter target
  _Float16* P   = (_Float16*)(ws + nA * 4);              // 128 MB power (ping)
  _Float16* Q   = (_Float16*)(ws + nA * 4 + nA * 2);     // 128 MB power (pong)
  _Float16* h16 = (_Float16*)(ws + nA * 8);              // 4 MB features f16

  k_zero_f32<<<2048, 256, 0, stream>>>(Af, (long)nA);
  k_scatter<<<(N_EDGES + 255) / 256, 256, 0, stream>>>(src, dst, e, Af);
  k_normalize<<<N_NODES, 256, 0, stream>>>(Af, P);
  k_f32_to_f16<<<2048, 256, 0, stream>>>(h, h16, (long)N_NODES * D_FEAT);

  const dim3 gSq(N_NODES / BN, N_NODES / BM);  // 64 x 64 blocks
  const dim3 gH(D_FEAT / BN, N_NODES / BM);    // 2  x 64 blocks
  const float invf[6] = {1.0f, 1.0f, 0.5f, 1.0f / 6.0f, 1.0f / 24.0f, 1.0f / 120.0f};

  // out = a @ h
  k_gemm_f16<false><<<gH, 256, 0, stream>>>(P, h16, out, D_FEAT, N_NODES, 1.0f, 0);

  _Float16* cur = P;
  _Float16* nxt = Q;
  for (int i = 1; i < 6; ++i) {
    // p = p @ p   (compute-dominant: 2*8192^3 flops on v_wmma_f32_16x16x32_f16)
    k_gemm_f16<true><<<gSq, 256, 0, stream>>>(cur, cur, nxt, N_NODES, N_NODES, 1.0f, 0);
    // out += p / i! @ h
    k_gemm_f16<false><<<gH, 256, 0, stream>>>(nxt, h16, out, D_FEAT, N_NODES, invf[i], 1);
    _Float16* tmp = cur; cur = nxt; nxt = tmp;
  }
}
